// RNNLM_Model_41764261987089
// MI455X (gfx1250) — matrix operand order repaired and checked by reading, not compiled, not run on hardware
//
#include <hip/hip_runtime.h>

#define VOCAB 32000
#define EMB   1024
#define HD    1024
#define TT    64
#define BB    32

typedef __attribute__((ext_vector_type(16))) __bf16          bf16x16;
typedef __attribute__((ext_vector_type(16))) unsigned short  u16x16;
typedef __attribute__((ext_vector_type(8)))  unsigned short  u16x8;
typedef __attribute__((ext_vector_type(8)))  float           f32x8;

__device__ __forceinline__ unsigned short f2bf(float f) {
    union { float f; unsigned u; } v; v.f = f;
    unsigned r = v.u + 0x7FFFu + ((v.u >> 16) & 1u);   // round-to-nearest-even
    return (unsigned short)(r >> 16);
}

// A fragment: 16x32 bf16, row-major source [*, lda].
// ISA layout: lane<16 -> row=lane, K = {0..7, 16..23}; lane>=16 -> row=lane-16, K = {8..15, 24..31}
__device__ __forceinline__ bf16x16 load_a_frag(const unsigned short* A, int lda,
                                               int row_base, int k0, int lane) {
    int row  = row_base + (lane & 15);
    int koff = (lane >> 4) << 3;
    const unsigned short* p = A + (long long)row * lda + k0 + koff;
    u16x8 lo = *(const u16x8*)p;
    u16x8 hi = *(const u16x8*)(p + 16);
    u16x16 t;
#pragma unroll
    for (int i = 0; i < 8; ++i) { t[i] = lo[i]; t[i + 8] = hi[i]; }
    return __builtin_bit_cast(bf16x16, t);
}

// B fragment: 32x16 bf16 from TRANSPOSED storage Bt[n][k] (ldb = K dim).
// ISA layout: lane<16 -> col=lane, K = {0..7, 16..23}; lane>=16 -> col=lane-16, K = {8..15, 24..31} (pairs per VGPR)
__device__ __forceinline__ bf16x16 load_bt_frag(const unsigned short* Bt, int ldb,
                                                int n_base, int k0, int lane) {
    int n    = n_base + (lane & 15);
    int koff = (lane >> 4) << 3;
    const unsigned short* p = Bt + (long long)n * ldb + k0 + koff;
    u16x8 lo = *(const u16x8*)p;
    u16x8 hi = *(const u16x8*)(p + 16);
    u16x16 t;
#pragma unroll
    for (int i = 0; i < 8; ++i) { t[i] = lo[i]; t[i + 8] = hi[i]; }
    return __builtin_bit_cast(bf16x16, t);
}

__device__ __forceinline__ f32x8 wmma_bf16(bf16x16 a, bf16x16 b, f32x8 c) {
    return __builtin_amdgcn_wmma_f32_16x16x32_bf16(false, a, false, b, (short)0, c, false, false);
}

// ---------- prep kernels ----------

// f32 [rows,cols] -> bf16 transposed [cols,rows]; rows,cols multiples of 32. block(32,8)
__global__ void transpose_bf16_tiled(const float* __restrict__ in,
                                     unsigned short* __restrict__ out,
                                     int rows, int cols) {
    __shared__ unsigned short tile[32][33];
    int c0 = blockIdx.x * 32, r0 = blockIdx.y * 32;
    int tx = threadIdx.x, ty = threadIdx.y;
#pragma unroll
    for (int i = ty; i < 32; i += 8)
        tile[i][tx] = f2bf(in[(long long)(r0 + i) * cols + c0 + tx]);
    __syncthreads();
#pragma unroll
    for (int i = ty; i < 32; i += 8)
        out[(long long)(c0 + i) * rows + r0 + tx] = tile[tx][i];
}

__global__ void to_bf16_kernel(const float* __restrict__ in,
                               unsigned short* __restrict__ out, int n) {
    int i = blockIdx.x * blockDim.x + threadIdx.x;
    if (i < n) out[i] = f2bf(in[i]);
}

// Gather embedding rows: row r = t*B + b  ->  emb[input_x[b,t]] as bf16
__global__ void gather_emb_bf16(const int* __restrict__ idx,
                                const float* __restrict__ emb,
                                unsigned short* __restrict__ Xb) {
    int r = blockIdx.x;              // 0..T*B-1
    int t = r / BB, b = r % BB;
    int v = idx[b * TT + t];
    const float* src = emb + (long long)v * EMB;
    unsigned short* dst = Xb + (long long)r * EMB;
    for (int i = threadIdx.x; i < EMB; i += blockDim.x) dst[i] = f2bf(src[i]);
}

// ---------- WMMA GEMM kernels ----------

// pre[T*B, HD] = X[T*B, EMB] @ I[EMB, HD] + b1   (It transposed bf16)
// grid(HD/16, T*B/16), block 32
__global__ void xi_gemm(const unsigned short* __restrict__ Xb,
                        const unsigned short* __restrict__ It,
                        const float* __restrict__ b1,
                        float* __restrict__ pre) {
    int lane = threadIdx.x;
    int nb = blockIdx.x * 16, mb = blockIdx.y * 16;
    f32x8 acc = {};
    for (int k0 = 0; k0 < EMB; k0 += 32) {
        bf16x16 a = load_a_frag(Xb, EMB, mb, k0, lane);
        bf16x16 b = load_bt_frag(It, EMB, nb, k0, lane);
        acc = wmma_bf16(a, b, acc);
    }
    int col = nb + (lane & 15);
    int rb  = mb + ((lane >> 4) << 3);
    float bias = b1[col];
#pragma unroll
    for (int i = 0; i < 8; ++i)
        pre[(long long)(rb + i) * HD + col] = acc[i] + bias;
}

// h_t = sigmoid(h_{t-1} @ H + pre_t);  M=B=32, N=K=HD.  grid(HD/16, 2), block 32
__global__ void rnn_step(const unsigned short* __restrict__ hprev,
                         const unsigned short* __restrict__ Ht,
                         const float* __restrict__ pre_t,
                         unsigned short* __restrict__ hout,
                         float* __restrict__ final_out) {
    int lane = threadIdx.x;
    int nb = blockIdx.x * 16, mb = blockIdx.y * 16;
    f32x8 acc = {};
    for (int k0 = 0; k0 < HD; k0 += 32) {
        bf16x16 a = load_a_frag(hprev, HD, mb, k0, lane);
        bf16x16 b = load_bt_frag(Ht, HD, nb, k0, lane);
        acc = wmma_bf16(a, b, acc);
    }
    int col = nb + (lane & 15);
    int rb  = mb + ((lane >> 4) << 3);
#pragma unroll
    for (int i = 0; i < 8; ++i) {
        long long o = (long long)(rb + i) * HD + col;
        float z = acc[i] + pre_t[o];
        float s = 1.0f / (1.0f + __expf(-z));
        hout[o] = f2bf(s);
        if (final_out) final_out[o] = s;
    }
}

// logits[T*B, V] = hs @ U + b2.  2x4 tile blocking per wave (32 rows x 64 cols).
// grid(V/256, T*B/32), block 128 (4 waves; wave w covers cols [bx*256+w*64, +64))
__global__ void proj_gemm(const unsigned short* __restrict__ hs,
                          const unsigned short* __restrict__ Ut,
                          const float* __restrict__ b2,
                          float* __restrict__ out) {
    int lane = threadIdx.x & 31;
    int w    = threadIdx.x >> 5;
    int mb   = blockIdx.y * 32;
    int nw   = blockIdx.x * 256 + w * 64;
    f32x8 acc[2][4] = {};
    for (int k0 = 0; k0 < HD; k0 += 32) {
        bf16x16 a0 = load_a_frag(hs, HD, mb,      k0, lane);
        bf16x16 a1 = load_a_frag(hs, HD, mb + 16, k0, lane);
#pragma unroll
        for (int j = 0; j < 4; ++j) {
            bf16x16 b = load_bt_frag(Ut, HD, nw + j * 16, k0, lane);
            acc[0][j] = wmma_bf16(a0, b, acc[0][j]);
            acc[1][j] = wmma_bf16(a1, b, acc[1][j]);
        }
    }
#pragma unroll
    for (int j = 0; j < 4; ++j) {
        int col = nw + j * 16 + (lane & 15);
        float bias = b2[col];
#pragma unroll
        for (int m = 0; m < 2; ++m) {
            int rb = mb + m * 16 + ((lane >> 4) << 3);
#pragma unroll
            for (int i = 0; i < 8; ++i) {
                // logits are write-once streaming data (262 MB): NT store keeps
                // the Ut/hs working set resident in the 192 MB L2.
                __builtin_nontemporal_store(acc[m][j][i] + bias,
                                            out + (long long)(rb + i) * VOCAB + col);
            }
        }
    }
}

// ---------- launcher ----------
extern "C" void kernel_launch(void* const* d_in, const int* in_sizes, int n_in,
                              void* d_out, int out_size, void* d_ws, size_t ws_size,
                              hipStream_t stream) {
    (void)in_sizes; (void)n_in; (void)out_size; (void)ws_size;
    const int*   input_x = (const int*)  d_in[0];
    const float* h0f     = (const float*)d_in[1];
    const float* emb     = (const float*)d_in[2];
    const float* Hf      = (const float*)d_in[3];
    const float* If      = (const float*)d_in[4];
    const float* b1      = (const float*)d_in[5];
    const float* Uf      = (const float*)d_in[6];
    const float* b2      = (const float*)d_in[7];
    float* outF = (float*)d_out;

    char* ws = (char*)d_ws;
    size_t off = 0;
    auto take = [&](size_t bytes) {
        void* p = ws + off;
        off += (bytes + 255) & ~(size_t)255;
        return p;
    };
    unsigned short* Ht  = (unsigned short*)take((size_t)HD * HD * 2);        // H^T bf16 [n][k]
    unsigned short* It  = (unsigned short*)take((size_t)EMB * HD * 2);       // I^T bf16 [n][k]
    unsigned short* Ut  = (unsigned short*)take((size_t)HD * VOCAB * 2);     // U^T bf16 [v][k]
    unsigned short* Xb  = (unsigned short*)take((size_t)TT * BB * EMB * 2);  // gathered emb bf16
    float*          pre = (float*)take((size_t)TT * BB * HD * 4);            // x@I + b1
    unsigned short* hsb = (unsigned short*)take((size_t)TT * BB * HD * 2);   // hidden states bf16
    unsigned short* h0b = (unsigned short*)take((size_t)BB * HD * 2);        // initial state bf16

    dim3 tb(32, 8);
    transpose_bf16_tiled<<<dim3(HD / 32, HD / 32), tb, 0, stream>>>(Hf, Ht, HD, HD);
    transpose_bf16_tiled<<<dim3(HD / 32, EMB / 32), tb, 0, stream>>>(If, It, EMB, HD);
    transpose_bf16_tiled<<<dim3(VOCAB / 32, HD / 32), tb, 0, stream>>>(Uf, Ut, HD, VOCAB);

    to_bf16_kernel<<<(BB * HD + 255) / 256, 256, 0, stream>>>(h0f, h0b, BB * HD);
    gather_emb_bf16<<<TT * BB, 256, 0, stream>>>(input_x, emb, Xb);

    xi_gemm<<<dim3(HD / 16, TT * BB / 16), 32, 0, stream>>>(Xb, It, b1, pre);

    for (int t = 0; t < TT; ++t) {
        const unsigned short* hprev = (t == 0) ? h0b : (hsb + (size_t)(t - 1) * BB * HD);
        float* fin = (t == TT - 1) ? (outF + (size_t)TT * BB * VOCAB) : nullptr;
        rnn_step<<<dim3(HD / 16, BB / 16), 32, 0, stream>>>(
            hprev, Ht, pre + (size_t)t * BB * HD, hsb + (size_t)t * BB * HD, fin);
    }

    proj_gemm<<<dim3(VOCAB / 256, TT * BB / 32), 128, 0, stream>>>(hsb, Ut, b2, outF);
}
